// ConvNeXtV2Block_63007170233029
// MI455X (gfx1250) — matrix operand order repaired and checked by reading, not compiled, not run on hardware
//
#include <hip/hip_runtime.h>
#include <hip/hip_bf16.h>

typedef _Float16 v16h __attribute__((ext_vector_type(16)));
typedef float    v8f  __attribute__((ext_vector_type(8)));
typedef unsigned int u32x4 __attribute__((ext_vector_type(4)));
typedef int          i32x4 __attribute__((ext_vector_type(4)));
typedef int          i32x8 __attribute__((ext_vector_type(8)));

#define B_  32
#define C_  256
#define T_  4096
#define KS  9
#define H_  512
#define TT  64           // tokens per subtile
#define CV  (C_ + 1)     // conv LDS stride (floats)  -> conflict-free LN reads
#define CP  (C_ + 8)     // yln LDS stride (halves)   -> 16B-aligned rows
#define HP  (H_ + 8)     // pass-B LDS stride (halves) == TDM pad of 4 dwords/row

union Frag { v16h v; uint4 u[2]; };
union Pack8 { _Float16 h[8]; uint4 u; };

// ---------------------------------------------------------------------------
// TDM: issue a 2D tensor_load_to_lds.  tile = 512 halves x 64 rows,
// row stride in memory = 1024 B, LDS padded by 16 B per 1024 B (-> HP stride).
// ---------------------------------------------------------------------------
__device__ __forceinline__ void tdm_load_tile(const _Float16* gsrc, unsigned lds_byte) {
    unsigned long long ga = (unsigned long long)(uintptr_t)gsrc;
    u32x4 g0;
    g0.x = 1u;                                            // count=1, user desc
    g0.y = lds_byte;                                      // lds_addr
    g0.z = (unsigned)ga;                                  // global_addr[31:0]
    g0.w = (unsigned)((ga >> 32) & 0x01FFFFFFu) | (2u << 30); // addr[56:32] | type=2
    i32x8 g1;
    g1[0] = (int)((1u << 16) | (1u << 20) | (7u << 22) | (3u << 25));
            // data_size=2B, pad_enable, pad_interval=256dw, pad_amount=4dw
    g1[1] = (int)(512u << 16);    // tensor_dim0 = 512 (low 16 in bits 63:48)
    g1[2] = (int)(4096u << 16);   // tensor_dim0 hi=0 | tensor_dim1 = 4096
    g1[3] = (int)(512u << 16);    // tensor_dim1 hi=0 | tile_dim0 = 512
    g1[4] = 64;                   // tile_dim1 = 64, tile_dim2 = 0
    g1[5] = 512;                  // tensor_dim0_stride = 512 elements
    g1[6] = 0;
    g1[7] = 0;
    i32x4 z4 = {};
#if defined(__clang_major__) && (__clang_major__ >= 23)
    i32x8 z8 = {};
    __builtin_amdgcn_tensor_load_to_lds(g0, g1, z4, z4, z8, 0);
#else
    __builtin_amdgcn_tensor_load_to_lds(g0, g1, z4, z4, 0);
#endif
}

// ---------------------------------------------------------------------------
// Kernel 0: convert weights fp32 -> f16 into workspace, zero sumsq.
// ---------------------------------------------------------------------------
__global__ __launch_bounds__(256) void prep_kernel(
    const float* __restrict__ w1f, const float* __restrict__ w2f,
    _Float16* __restrict__ w1h, _Float16* __restrict__ w2h,
    float* __restrict__ sumsq)
{
    int i = blockIdx.x * 256 + threadIdx.x;
    if (i < H_ * C_) {
        w1h[i] = (_Float16)w1f[i];
        w2h[i] = (_Float16)w2f[i];
    }
    if (i < B_ * H_) sumsq[i] = 0.0f;
}

// ---------------------------------------------------------------------------
// Kernel 0b: fold GRN beta into per-channel bias:
//   bias2c[c] = pw2_b[c] + sum_h pw2_w[c,h] * grn_beta[h]   (b-independent)
// ---------------------------------------------------------------------------
__global__ __launch_bounds__(256) void bias2c_kernel(
    const float* __restrict__ pw2_w, const float* __restrict__ pw2_b,
    const float* __restrict__ grn_beta, float* __restrict__ bias2c)
{
    int c = threadIdx.x;
    float s = pw2_b[c];
    const float* row = pw2_w + (size_t)c * H_;
    for (int h = 0; h < H_; ++h) s += row[h] * grn_beta[h];
    bias2c[c] = s;
}

// ---------------------------------------------------------------------------
// Kernel 1: dwconv(edge pad) -> channel LN -> PW1 (WMMA f16) -> GELU
//           -> store y1 f16 [b][t][h], accumulate sum of squares per (b,h).
// ---------------------------------------------------------------------------
__global__ __launch_bounds__(256) void passA_kernel(
    const float* __restrict__ x,
    const float* __restrict__ dw_w, const float* __restrict__ dw_b,
    const float* __restrict__ ln_w, const float* __restrict__ ln_b,
    const _Float16* __restrict__ w1, const float* __restrict__ pw1_b,
    _Float16* __restrict__ y1, float* __restrict__ sumsq)
{
    __shared__ float     conv[TT * CV];     // 65,792 B
    __shared__ _Float16  yln[TT * CP];      // 33,792 B, layout [t][c]
    __shared__ float     mu_s[TT], rs_s[TT];
    __shared__ float     bias1[H_];

    const int tid = threadIdx.x;
    const int b   = blockIdx.x / (T_ / TT);
    const int t0  = (blockIdx.x % (T_ / TT)) * TT;

    for (int i = tid; i < H_; i += 256) bias1[i] = pw1_b[i];

    // --- depthwise conv with replicate padding (fp32) ---
    const float* xb = x + (size_t)b * C_ * T_;
    for (int i = 0; i < (C_ * TT) / 256; ++i) {
        int e = i * 256 + tid;
        int c = e >> 6;          // e / TT
        int t = e & (TT - 1);
        float acc = dw_b[c];
        const float* xr = xb + (size_t)c * T_;
#pragma unroll
        for (int k = 0; k < KS; ++k) {
            int tt = t0 + t + k - (KS / 2);
            tt = tt < 0 ? 0 : (tt >= T_ ? T_ - 1 : tt);
            acc += dw_w[c * KS + k] * xr[tt];
        }
        conv[t * CV + c] = acc;
    }
    __syncthreads();

    // --- LayerNorm over C per token (threads 0..63, one token each) ---
    if (tid < TT) {
        const float* row = &conv[tid * CV];
        float s = 0.0f;
        for (int c = 0; c < C_; ++c) s += row[c];
        float mu = s * (1.0f / C_);
        float v = 0.0f;
        for (int c = 0; c < C_; ++c) { float d = row[c] - mu; v += d * d; }
        mu_s[tid] = mu;
        rs_s[tid] = rsqrtf(v * (1.0f / C_) + 1e-5f);
    }
    __syncthreads();

    // --- normalize + affine -> f16 B-operand tile [t][c] ---
    for (int i = 0; i < (C_ * TT) / 256; ++i) {
        int e = i * 256 + tid;
        int c = e >> 6;
        int t = e & (TT - 1);
        float v = (conv[t * CV + c] - mu_s[t]) * rs_s[t] * ln_w[c] + ln_b[c];
        yln[t * CP + c] = (_Float16)v;
    }
    __syncthreads();

    // --- WMMA GEMM: D(H x TT) = pw1_w(H x C) * yln(C x TT) ---
    const int wave = tid >> 5;
    const int lane = tid & 31;
    const int g    = lane >> 4;
    const int r    = lane & 15;

    for (int mi = 0; mi < 4; ++mi) {
        const int m0 = (wave * 4 + mi) * 16;           // h tile base
        v16h afr[8];
#pragma unroll
        for (int k = 0; k < 8; ++k) {                  // preload A frags (row-major f16)
            const _Float16* p = w1 + (size_t)(m0 + r) * C_ + k * 32 + 8 * g;
            Frag u; u.u[0] = *(const uint4*)p; u.u[1] = *(const uint4*)(p + 16);
            afr[k] = u.v;
        }
        for (int n = 0; n < 4; ++n) {
            const int nb = n * 16;
            v8f acc = {};
#pragma unroll
            for (int k = 0; k < 8; ++k) {
                const _Float16* p = &yln[(nb + r) * CP + k * 32 + 8 * g];
                Frag u; u.u[0] = *(const uint4*)p; u.u[1] = *(const uint4*)(p + 16);
                acc = __builtin_amdgcn_wmma_f32_16x16x32_f16(
                    false, afr[k], false, u.v, (short)0, acc, false, false);
            }
            // epilogue: bias + exact GELU, pack f16, store, sumsq reduce
            const int t  = t0 + nb + r;
            const int h0 = m0 + 8 * g;
            Pack8 P;
            float gl[8];
#pragma unroll
            for (int rr = 0; rr < 8; ++rr) {
                float v  = acc[rr] + bias1[h0 + rr];
                float ge = 0.5f * v * (1.0f + erff(v * 0.70710678118f));
                gl[rr]   = ge;
                P.h[rr]  = (_Float16)ge;
            }
            *(uint4*)(y1 + ((size_t)b * T_ + t) * H_ + h0) = P.u;
#pragma unroll
            for (int rr = 0; rr < 8; ++rr) {
                float s = gl[rr] * gl[rr];
                s += __shfl_xor(s, 1, 16);
                s += __shfl_xor(s, 2, 16);
                s += __shfl_xor(s, 4, 16);
                s += __shfl_xor(s, 8, 16);
                if (r == 0) atomicAdd(&sumsq[b * H_ + h0 + rr], s);
            }
        }
    }
}

// ---------------------------------------------------------------------------
// Kernel 2: GRN -> per-(b,h) scale = gamma*nx + 1  (beta folded into bias2c).
// ---------------------------------------------------------------------------
__global__ __launch_bounds__(256) void grn_kernel(
    const float* __restrict__ sumsq, const float* __restrict__ gamma,
    float* __restrict__ sA)
{
    __shared__ float red[256];
    const int b = blockIdx.x, tid = threadIdx.x;
    float g0 = sqrtf(sumsq[b * H_ + tid]);
    float g1 = sqrtf(sumsq[b * H_ + 256 + tid]);
    red[tid] = g0 + g1;
    __syncthreads();
    for (int s = 128; s > 0; s >>= 1) {
        if (tid < s) red[tid] += red[tid + s];
        __syncthreads();
    }
    float inv = 1.0f / (red[0] * (1.0f / H_) + 1e-6f);
    sA[b * H_ + tid]       = gamma[tid]       * (g0 * inv) + 1.0f;
    sA[b * H_ + 256 + tid] = gamma[256 + tid] * (g1 * inv) + 1.0f;
}

// ---------------------------------------------------------------------------
// Kernel 2b: per-batch scaled PW2 weights:  w2b[b][c][h] = w2[c][h] * sA[b][h]
// 8 MB f16 total -> stays L2-resident for passB.
// ---------------------------------------------------------------------------
__global__ __launch_bounds__(256) void scaleW2_kernel(
    const _Float16* __restrict__ w2h, const float* __restrict__ sA,
    _Float16* __restrict__ w2b)
{
    int e  = blockIdx.x * 256 + threadIdx.x;     // chunk of 8 halves
    int b  = e >> 14;                            // / (C_*H_/8)
    int r  = e & 16383;
    int c  = r >> 6;
    int h0 = (r & 63) * 8;
    Pack8 L, S;
    L.u = *(const uint4*)(w2h + (size_t)c * H_ + h0);
#pragma unroll
    for (int j = 0; j < 8; ++j)
        S.h[j] = (_Float16)((float)L.h[j] * sA[b * H_ + h0 + j]);
    *(uint4*)(w2b + ((size_t)(b * C_ + c)) * H_ + h0) = S.u;
}

// ---------------------------------------------------------------------------
// Kernel 3: TDM double-buffered y1 tiles -> PW2 (WMMA f16, per-b weights)
//           -> + bias2c + residual.
// ---------------------------------------------------------------------------
__device__ __forceinline__ void pw2_subtile(
    const _Float16* __restrict__ bBuf,   // LDS [TT][HP]
    const _Float16* __restrict__ w2b,    // per-b weights C_ x H_
    const float* __restrict__ bias2c_l,  // LDS
    const float* __restrict__ x, float* __restrict__ out,
    int b, int tbase, int wave, int lane)
{
    const int g = lane >> 4;
    const int r = lane & 15;
    for (int i = 0; i < 8; ++i) {
        const int tile = wave * 8 + i;
        const int m0 = (tile >> 2) * 16;   // c tile base
        const int n0 = (tile & 3) * 16;    // t tile base
        v8f acc = {};
#pragma unroll
        for (int k = 0; k < 16; ++k) {
            Frag ua, ub;
            const _Float16* pa = w2b + (size_t)(m0 + r) * H_ + k * 32 + 8 * g;
            ua.u[0] = *(const uint4*)pa; ua.u[1] = *(const uint4*)(pa + 16);
            const _Float16* pb = &bBuf[(n0 + r) * HP + k * 32 + 8 * g];
            ub.u[0] = *(const uint4*)pb; ub.u[1] = *(const uint4*)(pb + 16);
            acc = __builtin_amdgcn_wmma_f32_16x16x32_f16(
                false, ua.v, false, ub.v, (short)0, acc, false, false);
        }
        const int t  = tbase + n0 + r;
        const int c0 = m0 + 8 * g;
#pragma unroll
        for (int rr = 0; rr < 8; ++rr) {
            size_t idx = (size_t)b * C_ * T_ + (size_t)(c0 + rr) * T_ + t;
            out[idx] = acc[rr] + bias2c_l[c0 + rr] + x[idx];
        }
    }
}

__global__ __launch_bounds__(256) void passB_kernel(
    const _Float16* __restrict__ y1, const _Float16* __restrict__ w2b,
    const float* __restrict__ bias2c,
    const float* __restrict__ x, float* __restrict__ out)
{
    __shared__ _Float16 bBuf[2][TT * HP];   // 2 x 66,560 B, layout [t][h]
    __shared__ float bias2_l[C_];

    const int tid = threadIdx.x;
    const int b   = blockIdx.x / (T_ / (2 * TT));
    const int t0  = (blockIdx.x % (T_ / (2 * TT))) * (2 * TT);

    for (int i = tid; i < C_; i += 256) bias2_l[i] = bias2c[i];

    const bool loader = (tid < 32);          // wave 0 issues DMA (EXEC-independent op)
    if (loader) {
        tdm_load_tile(y1 + ((size_t)b * T_ + t0) * H_,
                      (unsigned)(uintptr_t)&bBuf[0][0]);
        tdm_load_tile(y1 + ((size_t)b * T_ + t0 + TT) * H_,
                      (unsigned)(uintptr_t)&bBuf[1][0]);
    }

    const _Float16* wb = w2b + (size_t)b * C_ * H_;
    const int wave = tid >> 5;
    const int lane = tid & 31;

    if (loader) __builtin_amdgcn_s_wait_tensorcnt(1);   // subtile 0 landed
    __syncthreads();
    pw2_subtile(&bBuf[0][0], wb, bias2_l, x, out, b, t0, wave, lane);

    if (loader) __builtin_amdgcn_s_wait_tensorcnt(0);   // subtile 1 landed
    __syncthreads();
    pw2_subtile(&bBuf[1][0], wb, bias2_l, x, out, b, t0 + TT, wave, lane);
}

// ---------------------------------------------------------------------------
extern "C" void kernel_launch(void* const* d_in, const int* in_sizes, int n_in,
                              void* d_out, int out_size, void* d_ws, size_t ws_size,
                              hipStream_t stream) {
    const float* x     = (const float*)d_in[0];
    const float* dw_w  = (const float*)d_in[1];
    const float* dw_b  = (const float*)d_in[2];
    const float* ln_w  = (const float*)d_in[3];
    const float* ln_b  = (const float*)d_in[4];
    const float* pw1_w = (const float*)d_in[5];
    const float* pw1_b = (const float*)d_in[6];
    const float* grn_g = (const float*)d_in[7];
    const float* grn_b = (const float*)d_in[8];
    const float* pw2_w = (const float*)d_in[9];
    const float* pw2_b = (const float*)d_in[10];
    float* out = (float*)d_out;

    // workspace layout
    char* ws = (char*)d_ws;
    const size_t y1_bytes  = (size_t)B_ * T_ * H_ * sizeof(_Float16);  // 128 MB
    const size_t w_bytes   = (size_t)H_ * C_ * sizeof(_Float16);       // 256 KB
    const size_t bh_bytes  = (size_t)B_ * H_ * sizeof(float);          // 64 KB
    _Float16* y1     = (_Float16*)ws;
    _Float16* w1h    = (_Float16*)(ws + y1_bytes);
    _Float16* w2h    = (_Float16*)(ws + y1_bytes + w_bytes);
    float*    sumsq  = (float*)(ws + y1_bytes + 2 * w_bytes);
    float*    sAv    = (float*)(ws + y1_bytes + 2 * w_bytes + bh_bytes);
    _Float16* w2b    = (_Float16*)(ws + y1_bytes + 2 * w_bytes + 2 * bh_bytes);
    float*    bias2c = (float*)(ws + y1_bytes + 2 * w_bytes + 2 * bh_bytes
                                + (size_t)B_ * C_ * H_ * sizeof(_Float16));

    prep_kernel<<<(H_ * C_ + 255) / 256, 256, 0, stream>>>(pw1_w, pw2_w, w1h, w2h, sumsq);
    bias2c_kernel<<<1, 256, 0, stream>>>(pw2_w, pw2_b, grn_b, bias2c);
    passA_kernel<<<B_ * (T_ / TT), 256, 0, stream>>>(
        x, dw_w, dw_b, ln_w, ln_b, w1h, pw1_b, y1, sumsq);
    grn_kernel<<<B_, 256, 0, stream>>>(sumsq, grn_g, sAv);
    scaleW2_kernel<<<(B_ * C_ * H_ / 8) / 256, 256, 0, stream>>>(w2h, sAv, w2b);
    passB_kernel<<<B_ * (T_ / (2 * TT)), 256, 0, stream>>>(y1, w2b, bias2c, x, out);
}
